// SiameseGNN_GIN_50886772523364
// MI455X (gfx1250) — matrix operand order repaired and checked by reading, not compiled, not run on hardware
//
#include <hip/hip_runtime.h>
#include <hip/hip_bf16.h>

// Problem constants (match reference)
#define N1_NODES 200000
#define N2_NODES 199
#define BGR      8
#define NPG      25000
#define KSEL     50
#define E1_EDGES 2000000LL
#define E2_EDGES 3184LL

typedef __attribute__((ext_vector_type(16))) __bf16   v16bf;
typedef __attribute__((ext_vector_type(8)))  float    v8f;
typedef __attribute__((ext_vector_type(8)))  unsigned v8ui;

// ---------------------------------------------------------- bf16 pack helpers
static __device__ __forceinline__ unsigned bfbits(float f) {
  union { float f; unsigned u; } c; c.f = f;
  return (c.u + 0x7FFFu + ((c.u >> 16) & 1u)) >> 16;   // RNE
}
static __device__ __forceinline__ unsigned bfpack(float lo, float hi) {
  return bfbits(lo) | (bfbits(hi) << 16);
}

// ------------------------------------------------- Tensor Data Mover (gfx1250)
#ifndef USE_TDM
#define USE_TDM 1
#endif
#if USE_TDM && defined(__has_builtin)
#if __has_builtin(__builtin_amdgcn_tensor_load_to_lds) && __has_builtin(__builtin_amdgcn_s_wait_tensorcnt)
#define HAVE_TDM 1
#endif
#endif
#ifndef HAVE_TDM
#define HAVE_TDM 0
#endif

#if HAVE_TDM
typedef unsigned int u32x4 __attribute__((ext_vector_type(4)));
typedef int          i32x8 __attribute__((ext_vector_type(8)));
typedef int          i32x4 __attribute__((ext_vector_type(4)));

// 2-D TDM load: tile_rows x dim0 f32 elements, row stride dim0, OOB rows -> 0.
// D# bit layout per CDNA5 ISA ch.8 (group0: count/lds/global/type, group1: dims).
// This toolchain exposes the 6-arg builtin: (g0, g1, g2, g3, g_extra, cpol).
static __device__ __forceinline__ void tdm_load_2d(unsigned ldsOff, const void* gaddr,
                                                   unsigned dim0, unsigned rows_avail,
                                                   unsigned tile_rows) {
  unsigned long long ga = (unsigned long long)gaddr;
  u32x4 g0;
  g0[0] = 1u;                                              // count=1, user mode
  g0[1] = ldsOff;                                          // lds_addr (bytes)
  g0[2] = (unsigned)ga;                                    // global_addr[31:0]
  g0[3] = (unsigned)((ga >> 32) & 0x1FFFFFFu) | (2u << 30);// addr[56:32] | type=2
  i32x8 g1;
  g1[0] = (int)(2u << 16);                                 // data_size=4B, mask=0
  g1[1] = (int)(dim0 << 16);                               // tensor_dim0[15:0]
  g1[2] = (int)((dim0 >> 16) | (rows_avail << 16));        // dim0 hi | tensor_dim1 lo
  g1[3] = (int)((rows_avail >> 16) | (dim0 << 16));        // dim1 hi | tile_dim0
  g1[4] = (int)(tile_rows & 0xFFFFu);                      // tile_dim1 (tile_dim2=0)
  g1[5] = (int)dim0;                                       // tensor_dim0_stride lo
  g1[6] = 0;
  g1[7] = 0;
  i32x4 z4 = {0, 0, 0, 0};
  i32x8 z8 = {0, 0, 0, 0, 0, 0, 0, 0};
  __builtin_amdgcn_tensor_load_to_lds(g0, g1, z4, z4, z8, 0);
}
#endif

// ---------------------------------------------------------------- copy
__global__ void copy4_kernel(const float4* __restrict__ src, float4* __restrict__ dst,
                             long long n4) {
  long long i = (long long)blockIdx.x * blockDim.x + threadIdx.x;
  if (i < n4) dst[i] = src[i];
}

// ------------------------------------------------- edge scatter-add (GIN agg)
// wave-per-edge: lane handles a 4-float feature chunk (F=128 fixed)
__global__ void scatter_add_kernel(const float* __restrict__ src, const int* __restrict__ ei,
                                   float* __restrict__ dst, long long E) {
  long long e = (long long)blockIdx.x * 8 + (threadIdx.x >> 5);
  if (e >= E) return;
  const int c = (threadIdx.x & 31) << 2;
  const int s = ei[e];
  const int d = ei[E + e];
  const float4 v = *(const float4*)(src + (size_t)s * 128 + c);
  float* dp = dst + (size_t)d * 128 + c;
  unsafeAtomicAdd(dp + 0, v.x);
  unsafeAtomicAdd(dp + 1, v.y);
  unsafeAtomicAdd(dp + 2, v.z);
  unsafeAtomicAdd(dp + 3, v.w);
}

// --------------------------------------- weight pre-pack: W[K][N] f32 -> bf16x2
// Wp dword(k2,n) = { bf16(W[2k2][n]), bf16(W[2k2+1][n]) } -> exact B-fragment words
__global__ void pack_weights_kernel(const float* __restrict__ W, unsigned* __restrict__ Wp,
                                    int K2, int N) {
  int i = blockIdx.x * blockDim.x + threadIdx.x;
  if (i >= K2 * N) return;
  int k2 = i / N, n = i - k2 * N;
  Wp[i] = bfpack(W[(size_t)(2 * k2) * N + n], W[(size_t)(2 * k2 + 1) * N + n]);
}

// ---------------------- row pre-pack: X[M][64] f32 -> Xp[M][32] bf16x2 dwords
__global__ void pack_rows64_kernel(const float* __restrict__ X, unsigned* __restrict__ Xp,
                                   int M) {
  int i = blockIdx.x * blockDim.x + threadIdx.x;
  if (i >= M * 32) return;
  Xp[i] = bfpack(X[2 * i], X[2 * i + 1]);
}

// ====================== WMMA GEMM: C = act(A[MxK] * W[KxN] + bias) ===========
// 256 threads / 8 waves. Block tile: 32 (M) x N. A staged via TDM (OOB rows
// read 0) into LDS f32, converted once to pair-packed bf16. Wave (w) owns
// m-subtile (w&1)*16 and TPW = N/64 n-tiles starting at (w>>1)*TPW.
// NOTE: C must be padded to gridDim.x*32 rows (stores are unpredicated).
template <int K, int N>
__global__ void __launch_bounds__(256)
wmma_gemm_t(const float* __restrict__ A, const unsigned* __restrict__ Wp,
            const float* __restrict__ bias, float* __restrict__ C, int M, int act) {
  constexpr int TPW = N / 64;       // accum tiles per wave (1 or 2)
  constexpr int K2  = K / 2;
  constexpr int LBS = K2 + 1;       // padded LDS row stride (bank-conflict free)
  const int tid  = threadIdx.x;
  const int wave = tid >> 5;
  const int lane = tid & 31;
  const int half = lane >> 4;
  const int l16  = lane & 15;
  const int msub = (wave & 1) << 4;
  const int ntb  = (wave >> 1) * TPW;
  const int m0   = blockIdx.x * 32;

  __shared__ float    ldsF[32 * K];
  __shared__ unsigned ldsB[32 * LBS];

#if HAVE_TDM
  if (wave == 0) {
    tdm_load_2d((unsigned)(size_t)(void*)ldsF, A + (size_t)m0 * K,
                (unsigned)K, (unsigned)(M - m0), 32u);
    __builtin_amdgcn_s_wait_tensorcnt(0);
  }
  __syncthreads();
  asm volatile("" ::: "memory");
#else
  for (int i = tid; i < 32 * K; i += 256) {
    int r = i / K, c = i - r * K;
    int row = m0 + r; if (row >= M) row = M - 1;
    ldsF[i] = A[(size_t)row * K + c];
  }
  __syncthreads();
#endif
  // f32 -> pair-packed bf16 (one pass, out of the inner loop)
  for (int i = tid; i < 32 * K2; i += 256) {
    int r = i / K2, k2 = i - r * K2;
    ldsB[r * LBS + k2] = bfpack(ldsF[r * K + 2 * k2], ldsF[r * K + 2 * k2 + 1]);
  }
  __syncthreads();

  v8f zero = {0.f, 0.f, 0.f, 0.f, 0.f, 0.f, 0.f, 0.f};
  v8f acc[TPW];
#pragma unroll
  for (int t = 0; t < TPW; ++t) acc[t] = zero;

  const int rowBase = (msub + l16) * LBS;
#pragma unroll
  for (int k0 = 0; k0 < K; k0 += 32) {
    // A fragment (16x32 bf16): VGPR v <- dword (k0/2 + half*8 + v)
    v8ui au;
#pragma unroll
    for (int v = 0; v < 8; ++v)
      au[v] = ldsB[rowBase + (k0 >> 1) + (half << 3) + v];
    v16bf a = __builtin_bit_cast(v16bf, au);
#pragma unroll
    for (int t = 0; t < TPW; ++t) {
      // B fragment (32x16 bf16): VGPR v <- Wp[(k0/2 + half*8 + v)*N + n]
      const int n = ((ntb + t) << 4) + l16;
      v8ui bu;
#pragma unroll
      for (int v = 0; v < 8; ++v)
        bu[v] = Wp[(size_t)((k0 >> 1) + (half << 3) + v) * N + n];
      v16bf b = __builtin_bit_cast(v16bf, bu);
      acc[t] = __builtin_amdgcn_wmma_f32_16x16x32_bf16(false, a, false, b,
                                                       (short)0, acc[t], false, false);
    }
  }
  // epilogue: straight-line stores, r*N folds into immediate offsets
#pragma unroll
  for (int t = 0; t < TPW; ++t) {
    const int col = ((ntb + t) << 4) + l16;
    const float bv = bias[col];
    float* Cp = C + (size_t)(m0 + msub + (half << 3)) * N + col;
#pragma unroll
    for (int r = 0; r < 8; ++r) {
      float v = acc[t][r] + bv;
      if (act) v = fmaxf(v, 0.f);
      Cp[(size_t)r * N] = v;
    }
  }
}

// =================== cdist via WMMA: sqrt(|a|^2 + |b|^2 - 2 a.b) =============
// 128 threads / 4 waves; block tile 16 (M) x 64 (N); A staged in LDS bf16,
// B = pre-packed out2 rows (contiguous K dwords). M multiple of 16.
__global__ void __launch_bounds__(128)
wmma_cdist_kernel(const float* __restrict__ A, const unsigned* __restrict__ Bp,
                  const float* __restrict__ na, const float* __restrict__ nb,
                  float* __restrict__ D, int M, int N) {
  const int tid  = threadIdx.x;
  const int wave = tid >> 5;
  const int lane = tid & 31;
  const int half = lane >> 4;
  const int l16  = lane & 15;
  const int ntiles = (N + 15) >> 4;
  const int nt = blockIdx.y * 4 + wave;
  const int m0 = blockIdx.x * 16;
  __shared__ float    ldsF[16 * 64];
  __shared__ unsigned ldsB[16 * 33];
  for (int i = tid; i < 16 * 64; i += 128) {
    int r = i >> 6, c = i & 63;
    int row = m0 + r; if (row >= M) row = M - 1;
    ldsF[i] = A[(size_t)row * 64 + c];
  }
  __syncthreads();
  for (int i = tid; i < 16 * 32; i += 128) {
    int r = i >> 5, k2 = i & 31;
    ldsB[r * 33 + k2] = bfpack(ldsF[(i << 1)], ldsF[(i << 1) + 1]);
  }
  __syncthreads();
  if (nt >= ntiles) return;                       // wave-uniform, after barriers
  int ncol = (nt << 4) + l16; if (ncol >= N) ncol = N - 1;
  const unsigned* Brow = Bp + (size_t)ncol * 32;
  const int rowBase = l16 * 33;
  v8f acc = {0.f, 0.f, 0.f, 0.f, 0.f, 0.f, 0.f, 0.f};
#pragma unroll
  for (int k0 = 0; k0 < 64; k0 += 32) {
    v8ui au, bu;
#pragma unroll
    for (int v = 0; v < 8; ++v) {
      au[v] = ldsB[rowBase + (k0 >> 1) + (half << 3) + v];
      bu[v] = Brow[(k0 >> 1) + (half << 3) + v];
    }
    v16bf a = __builtin_bit_cast(v16bf, au);
    v16bf b = __builtin_bit_cast(v16bf, bu);
    acc = __builtin_amdgcn_wmma_f32_16x16x32_bf16(false, a, false, b,
                                                  (short)0, acc, false, false);
  }
  const int col = (nt << 4) + l16;
  if (col < N) {
    const float  nbv = nb[col];
    const float* nap = na + m0 + (half << 3);
    float*       Dp  = D + (size_t)(m0 + (half << 3)) * N + col;
#pragma unroll
    for (int r = 0; r < 8; ++r) {
      float sq = nap[r] + nbv - 2.f * acc[r];
      Dp[(size_t)r * N] = __builtin_amdgcn_sqrtf(fmaxf(sq, 0.f));
    }
  }
}

// ------------------------------------------------------------- row |x|^2, K=64
__global__ void rownorm64_kernel(const float* __restrict__ x, float* __restrict__ nrm, int M) {
  int i = blockIdx.x * blockDim.x + threadIdx.x;
  if (i >= M) return;
  const float4* r = (const float4*)(x + (size_t)i * 64);
  float s = 0.f;
#pragma unroll
  for (int c = 0; c < 16; ++c) { float4 v = r[c]; s += v.x * v.x + v.y * v.y + v.z * v.z + v.w * v.w; }
  nrm[i] = s;
}

// --------------------------------------------------- extract sort keys (col 198)
__global__ void keys_kernel(const float* __restrict__ dmat, float* __restrict__ keys, int n) {
  int i = blockIdx.x * blockDim.x + threadIdx.x;
  if (i < n) keys[i] = dmat[(size_t)i * N2_NODES + (N2_NODES - 1)];
}

// --------------- sort pooling: top-50 rows by key desc (stable: low index on tie)
__global__ void __launch_bounds__(256)
sortpool_kernel(const float* __restrict__ dmat, float* __restrict__ keys,
                float* __restrict__ pooled) {
  __shared__ float sv[256];
  __shared__ int   si[256];
  __shared__ int   sel[KSEL];
  const int g = blockIdx.x, tid = threadIdx.x;
  float* kg = keys + g * NPG;
  for (int p = 0; p < KSEL; ++p) {
    float bv = -3.4e38f; int bi = 0x7fffffff;
    for (int i = tid; i < NPG; i += 256) {
      float v = kg[i];
      if (v > bv || (v == bv && i < bi)) { bv = v; bi = i; }
    }
    sv[tid] = bv; si[tid] = bi; __syncthreads();
    for (int s = 128; s > 0; s >>= 1) {
      if (tid < s) {
        if (sv[tid + s] > sv[tid] || (sv[tid + s] == sv[tid] && si[tid + s] < si[tid])) {
          sv[tid] = sv[tid + s]; si[tid] = si[tid + s];
        }
      }
      __syncthreads();
    }
    if (tid == 0) { sel[p] = si[0]; kg[si[0]] = -3.4e38f; }
    __syncthreads();
  }
  for (int t = tid; t < KSEL * N2_NODES; t += 256) {
    int r = t / N2_NODES, c = t - r * N2_NODES;
    pooled[(size_t)g * (KSEL * N2_NODES) + t] =
        dmat[(size_t)(g * NPG + sel[r]) * N2_NODES + c];
  }
}

// ----------------------- head: fc1+LN+relu, fc2+LN+relu, fc3+sigmoid (8 rows)
__global__ void __launch_bounds__(128)
head_kernel(const float* __restrict__ pooled,
            const float* __restrict__ W1, const float* __restrict__ b1,
            const float* __restrict__ g1, const float* __restrict__ be1,
            const float* __restrict__ W2, const float* __restrict__ b2,
            const float* __restrict__ g2, const float* __restrict__ be2,
            const float* __restrict__ W3, const float* __restrict__ b3,
            float* __restrict__ out) {
  __shared__ float h1[128];
  __shared__ float h2[64];
  __shared__ float red[128];
  const int g = blockIdx.x, tid = threadIdx.x;
  const float* p = pooled + (size_t)g * (KSEL * N2_NODES);
  float acc = b1[tid];
  for (int j = 0; j < KSEL * N2_NODES; ++j) acc = fmaf(p[j], W1[(size_t)j * 128 + tid], acc);
  red[tid] = acc; __syncthreads();
  for (int s = 64; s > 0; s >>= 1) { if (tid < s) red[tid] += red[tid + s]; __syncthreads(); }
  float mu = red[0] / 128.f; __syncthreads();
  float cen = acc - mu;
  red[tid] = cen * cen; __syncthreads();
  for (int s = 64; s > 0; s >>= 1) { if (tid < s) red[tid] += red[tid + s]; __syncthreads(); }
  float var = red[0] / 128.f; __syncthreads();
  h1[tid] = fmaxf(cen * rsqrtf(var + 1e-5f) * g1[tid] + be1[tid], 0.f);
  __syncthreads();
  float acc2 = 0.f;
  if (tid < 64) {
    acc2 = b2[tid];
    for (int j = 0; j < 128; ++j) acc2 = fmaf(h1[j], W2[(size_t)j * 64 + tid], acc2);
  }
  red[tid] = (tid < 64) ? acc2 : 0.f; __syncthreads();
  for (int s = 64; s > 0; s >>= 1) { if (tid < s) red[tid] += red[tid + s]; __syncthreads(); }
  float mu2 = red[0] / 64.f; __syncthreads();
  float cen2 = acc2 - mu2;
  red[tid] = (tid < 64) ? cen2 * cen2 : 0.f; __syncthreads();
  for (int s = 64; s > 0; s >>= 1) { if (tid < s) red[tid] += red[tid + s]; __syncthreads(); }
  float var2 = red[0] / 64.f; __syncthreads();
  if (tid < 64) h2[tid] = fmaxf(cen2 * rsqrtf(var2 + 1e-5f) * g2[tid] + be2[tid], 0.f);
  __syncthreads();
  red[tid] = (tid < 64) ? h2[tid] * W3[tid] : 0.f; __syncthreads();
  for (int s = 64; s > 0; s >>= 1) { if (tid < s) red[tid] += red[tid + s]; __syncthreads(); }
  if (tid == 0) {
    float z = red[0] + b3[0];
    out[g] = 1.f / (1.f + expf(-z));
  }
}

// ============================================================== launcher
extern "C" void kernel_launch(void* const* d_in, const int* in_sizes, int n_in,
                              void* d_out, int out_size, void* d_ws, size_t ws_size,
                              hipStream_t stream) {
  (void)in_sizes; (void)n_in; (void)out_size; (void)ws_size;
  const float* x1   = (const float*)d_in[0];
  const int*   ei1  = (const int*)d_in[1];
  const float* x2   = (const float*)d_in[3];
  const int*   ei2  = (const int*)d_in[4];
  const float* W1a  = (const float*)d_in[5];
  const float* b1a  = (const float*)d_in[6];
  const float* W1b  = (const float*)d_in[7];
  const float* b1b  = (const float*)d_in[8];
  const float* W2a  = (const float*)d_in[9];
  const float* b2a  = (const float*)d_in[10];
  const float* W2b  = (const float*)d_in[11];
  const float* b2b  = (const float*)d_in[12];
  const float* fc1W = (const float*)d_in[13];
  const float* fc1b = (const float*)d_in[14];
  const float* g1   = (const float*)d_in[15];
  const float* be1  = (const float*)d_in[16];
  const float* fc2W = (const float*)d_in[17];
  const float* fc2b = (const float*)d_in[18];
  const float* g2   = (const float*)d_in[19];
  const float* be2  = (const float*)d_in[20];
  const float* fc3W = (const float*)d_in[21];
  const float* fc3b = (const float*)d_in[22];
  float* out = (float*)d_out;

  char* ws = (char*)d_ws;
  size_t off = 0;
  auto carve = [&](size_t bytes) -> void* {
    void* p = ws + off;
    off += (bytes + 255) & ~(size_t)255;
    return p;
  };
  float*    BUF0  = (float*)carve((size_t)N1_NODES * 128 * 4);
  float*    BUF1  = (float*)carve((size_t)N1_NODES * 128 * 4);
  float*    DMAT  = (float*)carve((size_t)N1_NODES * N2_NODES * 4);
  float*    G2A   = (float*)carve((size_t)224 * 128 * 4);  // padded to 7 blocks * 32 rows
  float*    G2B   = (float*)carve((size_t)224 * 128 * 4);
  float*    G2C   = (float*)carve((size_t)224 * 128 * 4);
  float*    OUT2  = (float*)carve((size_t)224 * 64 * 4);
  float*    NRM1  = (float*)carve((size_t)N1_NODES * 4);
  float*    NRM2  = (float*)carve((size_t)224 * 4);
  float*    KEYS  = (float*)carve((size_t)BGR * NPG * 4);
  float*    POOL  = (float*)carve((size_t)BGR * KSEL * N2_NODES * 4);
  unsigned* W1aP  = (unsigned*)carve((size_t)64 * 128 * 4);
  unsigned* W1bP  = (unsigned*)carve((size_t)64 * 128 * 4);
  unsigned* W2aP  = (unsigned*)carve((size_t)64 * 64 * 4);
  unsigned* W2bP  = (unsigned*)carve((size_t)32 * 64 * 4);
  unsigned* OUT2P = (unsigned*)carve((size_t)224 * 32 * 4);

  auto copyN = [&](const float* s, float* d, long long nfloats) {
    long long n4 = nfloats >> 2;
    copy4_kernel<<<(unsigned)((n4 + 255) / 256), 256, 0, stream>>>((const float4*)s, (float4*)d, n4);
  };
  auto scatter = [&](const float* s, const int* ei, float* d, long long E) {
    scatter_add_kernel<<<(unsigned)((E + 7) / 8), 256, 0, stream>>>(s, ei, d, E);
  };
  auto pack = [&](const float* W, unsigned* Wp, int K2, int N) {
    int n = K2 * N;
    pack_weights_kernel<<<(n + 255) / 256, 256, 0, stream>>>(W, Wp, K2, N);
  };

  // pre-pack all shared GIN weights once per launch
  pack(W1a, W1aP, 64, 128);
  pack(W1b, W1bP, 64, 128);
  pack(W2a, W2aP, 64, 64);
  pack(W2b, W2bP, 32, 64);

  // ---- graph 1 GIN
  copyN(x1, BUF0, (long long)N1_NODES * 128);
  scatter(x1, ei1, BUF0, E1_EDGES);                                        // x + agg(x)
  wmma_gemm_t<128, 128><<<(N1_NODES + 31) / 32, 256, 0, stream>>>(BUF0, W1aP, b1a, BUF1, N1_NODES, 1);
  wmma_gemm_t<128, 128><<<(N1_NODES + 31) / 32, 256, 0, stream>>>(BUF1, W1bP, b1b, BUF0, N1_NODES, 1);
  copyN(BUF0, BUF1, (long long)N1_NODES * 128);
  scatter(BUF0, ei1, BUF1, E1_EDGES);                                      // h + agg(h)
  float* T2 = DMAT;                                                        // stage t2 in DMAT region
  wmma_gemm_t<128, 64><<<(N1_NODES + 31) / 32, 256, 0, stream>>>(BUF1, W2aP, b2a, T2, N1_NODES, 1);
  wmma_gemm_t<64, 64><<<(N1_NODES + 31) / 32, 256, 0, stream>>>(T2, W2bP, b2b, BUF0, N1_NODES, 1); // out1

  // ---- graph 2 GIN (shared weights)
  copyN(x2, G2A, (long long)N2_NODES * 128);
  scatter(x2, ei2, G2A, E2_EDGES);
  wmma_gemm_t<128, 128><<<(N2_NODES + 31) / 32, 256, 0, stream>>>(G2A, W1aP, b1a, G2B, N2_NODES, 1);
  wmma_gemm_t<128, 128><<<(N2_NODES + 31) / 32, 256, 0, stream>>>(G2B, W1bP, b1b, G2A, N2_NODES, 1);
  copyN(G2A, G2B, (long long)N2_NODES * 128);
  scatter(G2A, ei2, G2B, E2_EDGES);
  wmma_gemm_t<128, 64><<<(N2_NODES + 31) / 32, 256, 0, stream>>>(G2B, W2aP, b2a, G2C, N2_NODES, 1);
  wmma_gemm_t<64, 64><<<(N2_NODES + 31) / 32, 256, 0, stream>>>(G2C, W2bP, b2b, OUT2, N2_NODES, 1);

  // ---- cdist = sqrt(|a|^2 + |b|^2 - 2 a.b): WMMA dot + norm epilogue
  rownorm64_kernel<<<(N1_NODES + 255) / 256, 256, 0, stream>>>(BUF0, NRM1, N1_NODES);
  rownorm64_kernel<<<1, 256, 0, stream>>>(OUT2, NRM2, N2_NODES);
  pack_rows64_kernel<<<(N2_NODES * 32 + 255) / 256, 256, 0, stream>>>(OUT2, OUT2P, N2_NODES);
  {
    int ntiles = (N2_NODES + 15) / 16;                   // 13
    dim3 grid((unsigned)((N1_NODES + 15) / 16), (unsigned)((ntiles + 3) / 4));
    wmma_cdist_kernel<<<grid, 128, 0, stream>>>(BUF0, OUT2P, NRM1, NRM2, DMAT,
                                                N1_NODES, N2_NODES);
  }

  // ---- sort pooling + head
  keys_kernel<<<(BGR * NPG + 255) / 256, 256, 0, stream>>>(DMAT, KEYS, BGR * NPG);
  sortpool_kernel<<<BGR, 256, 0, stream>>>(DMAT, KEYS, POOL);
  head_kernel<<<BGR, 128, 0, stream>>>(POOL, fc1W, fc1b, g1, be1,
                                       fc2W, fc2b, g2, be2, fc3W, fc3b, out);
}
